// STGNN_69286412419322
// MI455X (gfx1250) — matrix operand order repaired and checked by reading, not compiled, not run on hardware
//
#include <hip/hip_runtime.h>
#include <hip/hip_bf16.h>

// ---------------------------------------------------------------------------
// STGNN: GRU(T=24,H=64) -> GCN -> GCN -> FC(12)   for MI455X / gfx1250
//
// Dense math uses v_wmma_f32_16x16x32_f16 (f16 in, f32 accum).
// Sparse aggregation uses float4 gathers + global_atomic_add_f32; the hot
// buffers (hw/agg, 12.8MB each) are L2-resident on the 192MB L2.
// ---------------------------------------------------------------------------

#define N_NODES 50000
#define T_WIN   24
#define HID     64
#define HOR_    12
#define NEDGE   1600000
#define NTILE   3125          // N_NODES / 16 (exact)

typedef __attribute__((ext_vector_type(16))) _Float16 v16h;
typedef __attribute__((ext_vector_type(8)))  _Float16 v8h;
typedef __attribute__((ext_vector_type(8)))  float    v8f;

__device__ __forceinline__ float sigm(float v) { return 1.0f / (1.0f + __expf(-v)); }

// ---------------------------------------------------------------------------
// Pre-swizzle weight matrices into the WMMA B-operand register image.
// B layout (16-bit, 32x16): lane l holds column n = l&15; halves i=0..15 hold
// K = kc*32 + (l<16 ? 0 : 8) + (i<8 ? i : i+8).  B[k][n] = W[colBase + n][k]  (gh = h @ W^T)
// Tiles: 0..23  w_hh  (12 col-blocks x 2 k-chunks)
//        24..31 gc1_w (4 x 2)
//        32..39 gc2_w (4 x 2)
//        40..41 fc_w  (1 x 2, cols 12..15 zero-padded)
// ---------------------------------------------------------------------------
__global__ void prep_swizzle(const float* __restrict__ w_hh,
                             const float* __restrict__ gc1_w,
                             const float* __restrict__ gc2_w,
                             const float* __restrict__ fc_w,
                             _Float16* __restrict__ wB) {
    int gid = blockIdx.x * 256 + threadIdx.x;
    if (gid >= 42 * 512) return;
    int tile   = gid >> 9;          // 0..41
    int within = gid & 511;
    int lane   = within >> 4;       // 0..31
    int i      = within & 15;       // half index
    int n      = lane & 15;         // column within tile

    const float* W; int cb, kc, ncols;
    if (tile < 24)      { W = w_hh;  int t2 = tile;      cb = t2 >> 1; kc = t2 & 1; ncols = 192; }
    else if (tile < 32) { W = gc1_w; int t2 = tile - 24; cb = t2 >> 1; kc = t2 & 1; ncols = 64;  }
    else if (tile < 40) { W = gc2_w; int t2 = tile - 32; cb = t2 >> 1; kc = t2 & 1; ncols = 64;  }
    else                { W = fc_w;  int t2 = tile - 40; cb = 0;       kc = t2 & 1; ncols = HOR_; }

    int k   = kc * 32 + ((lane < 16) ? 0 : 8) + ((i < 8) ? i : i + 8);
    int col = cb * 16 + n;
    float val = (col < ncols) ? W[col * 64 + k] : 0.0f;
    wB[gid] = (_Float16)val;
}

// ---------------------------------------------------------------------------
// GRU: one wave owns a 16-node tile; h kept in WMMA C-layout registers over
// all 24 steps.  Per step: 24x v_wmma_f32_16x16x32_f16 (gh = h @ w_hh^T),
// gate math in VALU, h converted to f16 into per-wave LDS for next step's A.
// ---------------------------------------------------------------------------
__global__ __launch_bounds__(256) void
gru_kernel(const float* __restrict__ x,
           const float* __restrict__ w_ih, const float* __restrict__ b_ih,
           const float* __restrict__ b_hh,
           const _Float16* __restrict__ wBhh,   // 24 tiles * 512 halves
           float* __restrict__ hout) {
    __shared__ __align__(32) _Float16 s_wB[24 * 512];   // 24KB swizzled w_hh
    __shared__ __align__(32) _Float16 s_h[8 * 16 * 64]; // 16KB, 2KB per wave

    // block-stage B tiles (uint4 copies)
    {
        const uint4* src = (const uint4*)wBhh;
        uint4*       dst = (uint4*)s_wB;
        for (int i = threadIdx.x; i < (24 * 512 * 2) / 16; i += 256) dst[i] = src[i];
    }
    __syncthreads();

    int lane = threadIdx.x & 31;
    int w    = threadIdx.x >> 5;
    int tile = blockIdx.x * 8 + w;
    if (tile >= NTILE) return;                 // wave-uniform exit

    int nb = tile * 16;
    int jl = lane & 15;
    int rb = (lane >= 16) ? 8 : 0;
    _Float16* myh = &s_h[w * 1024];

    for (int i = lane; i < 1024; i += 32) myh[i] = (_Float16)0.0f;   // h0 = 0
    __builtin_amdgcn_wave_barrier();

    // per-lane gate parameters: column j = tt*16 + (lane&15) within each 64-wide gate
    float wir[4], wiz[4], win_[4], bir[4], biz[4], bin_[4], bhr[4], bhz[4], bhn[4];
#pragma unroll
    for (int t = 0; t < 4; ++t) {
        int j = t * 16 + jl;
        wir[t] = w_ih[j];       wiz[t] = w_ih[64 + j];  win_[t] = w_ih[128 + j];
        bir[t] = b_ih[j];       biz[t] = b_ih[64 + j];  bin_[t] = b_ih[128 + j];
        bhr[t] = b_hh[j];       bhz[t] = b_hh[64 + j];  bhn[t] = b_hh[128 + j];
    }

    v8f hC[4] = {};   // hidden state, 4 C-tiles (cols tt*16+jl, rows v+rb)

    for (int t = 0; t < T_WIN; ++t) {
        // x_t for this lane's node column (lanes>=16 mirror 0..15)
        float xt = x[(size_t)(nb + jl) * T_WIN + t];

        // A tiles from LDS h16: lane row m=jl; halves = K[base..base+7] ++ K[base+16..base+23]
        v16h a[2];
#pragma unroll
        for (int kc = 0; kc < 2; ++kc) {
            int base = kc * 32 + ((lane < 16) ? 0 : 8);
            v8h lo = *(const v8h*)(&myh[jl * 64 + base]);
            v8h hi = *(const v8h*)(&myh[jl * 64 + base + 16]);
            a[kc] = __builtin_shufflevector(lo, hi, 0, 1, 2, 3, 4, 5, 6, 7,
                                                    8, 9, 10, 11, 12, 13, 14, 15);
        }

        // gh = h @ w_hh^T : 12 column tiles x 2 K-chunks
        v8f acc[12] = {};
#pragma unroll
        for (int cb = 0; cb < 12; ++cb) {
#pragma unroll
            for (int kc = 0; kc < 2; ++kc) {
                v16h b = *(const v16h*)(&s_wB[(cb * 2 + kc) * 512 + lane * 16]);
                acc[cb] = __builtin_amdgcn_wmma_f32_16x16x32_f16(
                    false, a[kc], false, b, (short)0, acc[cb], false, false);
            }
        }

        // broadcast x of row (v+rb) to all lanes of that half
        float xm[8];
#pragma unroll
        for (int v = 0; v < 8; ++v)
            xm[v] = __int_as_float(__builtin_amdgcn_ds_bpermute(
                        (v + rb) << 2, __float_as_int(xt)));

        // gates, elementwise in C-layout
#pragma unroll
        for (int tt = 0; tt < 4; ++tt) {
#pragma unroll
            for (int v = 0; v < 8; ++v) {
                float gr = xm[v] * wir[tt]  + bir[tt]  + acc[tt][v]     + bhr[tt];
                float gz = xm[v] * wiz[tt]  + biz[tt]  + acc[tt + 4][v] + bhz[tt];
                float gi = xm[v] * win_[tt] + bin_[tt];
                float r  = sigm(gr);
                float z  = sigm(gz);
                float nn = tanhf(gi + r * (acc[tt + 8][v] + bhn[tt]));
                hC[tt][v] = (1.0f - z) * nn + z * hC[tt][v];
            }
        }

        // write h back as f16 row-major for next step's A loads
#pragma unroll
        for (int tt = 0; tt < 4; ++tt)
#pragma unroll
            for (int v = 0; v < 8; ++v)
                myh[(v + rb) * 64 + tt * 16 + jl] = (_Float16)hC[tt][v];
        __builtin_amdgcn_wave_barrier();
    }

    // final hidden -> global [N,64]
#pragma unroll
    for (int tt = 0; tt < 4; ++tt)
#pragma unroll
        for (int v = 0; v < 8; ++v)
            hout[(size_t)(nb + v + rb) * 64 + tt * 16 + jl] = hC[tt][v];
}

// ---------------------------------------------------------------------------
// hw = h @ W^T : one wave per 16-node tile, 8 WMMAs (4 col tiles x 2 K-chunks)
// ---------------------------------------------------------------------------
__global__ __launch_bounds__(256) void
gcn_linear(const float* __restrict__ hin, const _Float16* __restrict__ wB,
           float* __restrict__ hw) {
    int lane = threadIdx.x & 31;
    int w    = threadIdx.x >> 5;
    int tile = blockIdx.x * 8 + w;
    if (tile >= NTILE) return;
    int nb = tile * 16, jl = lane & 15, rb = (lane >= 16) ? 8 : 0;

    const float* hrow = hin + (size_t)(nb + jl) * 64;
    v16h a[2];
#pragma unroll
    for (int kc = 0; kc < 2; ++kc) {
        int base = kc * 32 + ((lane < 16) ? 0 : 8);
        float4 f0 = *(const float4*)(hrow + base);
        float4 f1 = *(const float4*)(hrow + base + 4);
        float4 f2 = *(const float4*)(hrow + base + 16);
        float4 f3 = *(const float4*)(hrow + base + 20);
        float tmp[16] = {f0.x, f0.y, f0.z, f0.w, f1.x, f1.y, f1.z, f1.w,
                         f2.x, f2.y, f2.z, f2.w, f3.x, f3.y, f3.z, f3.w};
        v16h av;
#pragma unroll
        for (int i = 0; i < 16; ++i) av[i] = (_Float16)tmp[i];
        a[kc] = av;
    }

    v8f acc[4] = {};
#pragma unroll
    for (int cb = 0; cb < 4; ++cb)
#pragma unroll
        for (int kc = 0; kc < 2; ++kc) {
            v16h b = *(const v16h*)(wB + (cb * 2 + kc) * 512 + lane * 16);
            acc[cb] = __builtin_amdgcn_wmma_f32_16x16x32_f16(
                false, a[kc], false, b, (short)0, acc[cb], false, false);
        }

#pragma unroll
    for (int cb = 0; cb < 4; ++cb)
#pragma unroll
        for (int v = 0; v < 8; ++v)
            hw[(size_t)(nb + v + rb) * 64 + cb * 16 + jl] = acc[cb][v];
}

// ---------------------------------------------------------------------------
// GCN norm + aggregation
// ---------------------------------------------------------------------------
__global__ void deg_init(float* degf) {
    int i = blockIdx.x * 256 + threadIdx.x;
    if (i < N_NODES) degf[i] = 1.0f;             // self-loop contributes 1
}
__global__ void deg_edges(const int* __restrict__ ei, float* degf) {
    int e = blockIdx.x * 256 + threadIdx.x;
    if (e < NEDGE) atomicAdd(&degf[ei[NEDGE + e]], 1.0f);
}
__global__ void deg_fin(const float* __restrict__ degf, float* dinv) {
    int i = blockIdx.x * 256 + threadIdx.x;
    if (i < N_NODES) dinv[i] = rsqrtf(degf[i]);  // deg >= 1 always
}

// agg[n] = bias + hw[n]*dinv[n]^2   (self-loop term + bias init)
__global__ void gcn_self(const float* __restrict__ hw, const float* __restrict__ dinv,
                         const float* __restrict__ b, float* __restrict__ agg) {
    int g = blockIdx.x * 256 + threadIdx.x;
    if (g >= N_NODES * 64) return;
    int n = g >> 6, j = g & 63;
    float di = dinv[n];
    agg[g] = b[j] + hw[g] * di * di;
}

// edge scatter: 16 threads per edge, float4 gather from L2-resident hw
__global__ void gcn_agg(const float* __restrict__ hw, const int* __restrict__ ei,
                        const float* __restrict__ dinv, float* __restrict__ agg) {
    int g = blockIdx.x * 256 + threadIdx.x;
    int e = g >> 4;
    if (e >= NEDGE) return;
    int part = g & 15;
    int s = ei[e], d = ei[NEDGE + e];
    float nm = dinv[s] * dinv[d];
    const float4 v = *(const float4*)(hw + (size_t)s * 64 + part * 4);
    float* o = agg + (size_t)d * 64 + part * 4;
    atomicAdd(o + 0, v.x * nm);
    atomicAdd(o + 1, v.y * nm);
    atomicAdd(o + 2, v.z * nm);
    atomicAdd(o + 3, v.w * nm);
}

__global__ void relu_copy(const float* __restrict__ agg, float* __restrict__ h) {
    int g = blockIdx.x * 256 + threadIdx.x;
    if (g < N_NODES * 64) h[g] = fmaxf(agg[g], 0.0f);
}

// ---------------------------------------------------------------------------
// out = h @ fc_w^T + fc_b : one wave per tile, 2 WMMAs (cols 12..15 padded)
// ---------------------------------------------------------------------------
__global__ __launch_bounds__(256) void
fc_kernel(const float* __restrict__ hin, const _Float16* __restrict__ wBfc,
          const float* __restrict__ fc_b, float* __restrict__ out) {
    int lane = threadIdx.x & 31;
    int w    = threadIdx.x >> 5;
    int tile = blockIdx.x * 8 + w;
    if (tile >= NTILE) return;
    int nb = tile * 16, jl = lane & 15, rb = (lane >= 16) ? 8 : 0;

    const float* hrow = hin + (size_t)(nb + jl) * 64;
    v16h a[2];
#pragma unroll
    for (int kc = 0; kc < 2; ++kc) {
        int base = kc * 32 + ((lane < 16) ? 0 : 8);
        float4 f0 = *(const float4*)(hrow + base);
        float4 f1 = *(const float4*)(hrow + base + 4);
        float4 f2 = *(const float4*)(hrow + base + 16);
        float4 f3 = *(const float4*)(hrow + base + 20);
        float tmp[16] = {f0.x, f0.y, f0.z, f0.w, f1.x, f1.y, f1.z, f1.w,
                         f2.x, f2.y, f2.z, f2.w, f3.x, f3.y, f3.z, f3.w};
        v16h av;
#pragma unroll
        for (int i = 0; i < 16; ++i) av[i] = (_Float16)tmp[i];
        a[kc] = av;
    }

    v8f acc = {};
#pragma unroll
    for (int kc = 0; kc < 2; ++kc) {
        v16h b = *(const v16h*)(wBfc + kc * 512 + lane * 16);
        acc = __builtin_amdgcn_wmma_f32_16x16x32_f16(
            false, a[kc], false, b, (short)0, acc, false, false);
    }

    if (jl < HOR_) {
        float bias = fc_b[jl];
#pragma unroll
        for (int v = 0; v < 8; ++v)
            out[(size_t)(nb + v + rb) * HOR_ + jl] = acc[v] + bias;
    }
}

// ---------------------------------------------------------------------------
// launch
// ---------------------------------------------------------------------------
extern "C" void kernel_launch(void* const* d_in, const int* in_sizes, int n_in,
                              void* d_out, int out_size, void* d_ws, size_t ws_size,
                              hipStream_t stream) {
    const float* x     = (const float*)d_in[0];
    const int*   ei    = (const int*)  d_in[1];
    const float* w_ih  = (const float*)d_in[2];
    const float* w_hh  = (const float*)d_in[3];
    const float* b_ih  = (const float*)d_in[4];
    const float* b_hh  = (const float*)d_in[5];
    const float* gc1_w = (const float*)d_in[6];
    const float* gc1_b = (const float*)d_in[7];
    const float* gc2_w = (const float*)d_in[8];
    const float* gc2_b = (const float*)d_in[9];
    const float* fc_w  = (const float*)d_in[10];
    const float* fc_b  = (const float*)d_in[11];
    float* out = (float*)d_out;

    // workspace layout (floats): h | hw | agg | degf | dinv | wB(f16)
    float* wsf  = (float*)d_ws;
    float* h    = wsf;
    float* hw   = wsf + (size_t)1 * N_NODES * 64;
    float* agg  = wsf + (size_t)2 * N_NODES * 64;
    float* degf = wsf + (size_t)3 * N_NODES * 64;
    float* dinv = degf + N_NODES;
    _Float16* wB = (_Float16*)(dinv + N_NODES);   // byte offset 38.8MB, 32B aligned
    _Float16* wB_hh = wB;
    _Float16* wB_g1 = wB + 24 * 512;
    _Float16* wB_g2 = wB + 32 * 512;
    _Float16* wB_fc = wB + 40 * 512;

    const int TB = 256;
    const int gTiles = (NTILE + 7) / 8;           // 391 blocks, 8 waves each
    const int gN     = (N_NODES + TB - 1) / TB;   // 196
    const int gNH    = (N_NODES * 64) / TB;       // 12500
    const int gE     = NEDGE / TB;                // 6250
    const int gAgg   = (NEDGE * 16) / TB;         // 100000

    prep_swizzle<<<84, TB, 0, stream>>>(w_hh, gc1_w, gc2_w, fc_w, wB);

    deg_init <<<gN, TB, 0, stream>>>(degf);
    deg_edges<<<gE, TB, 0, stream>>>(ei, degf);
    deg_fin  <<<gN, TB, 0, stream>>>(degf, dinv);

    gru_kernel<<<gTiles, TB, 0, stream>>>(x, w_ih, b_ih, b_hh, wB_hh, h);

    // GCN layer 1
    gcn_linear<<<gTiles, TB, 0, stream>>>(h, wB_g1, hw);
    gcn_self  <<<gNH,    TB, 0, stream>>>(hw, dinv, gc1_b, agg);
    gcn_agg   <<<gAgg,   TB, 0, stream>>>(hw, ei, dinv, agg);
    relu_copy <<<gNH,    TB, 0, stream>>>(agg, h);

    // GCN layer 2
    gcn_linear<<<gTiles, TB, 0, stream>>>(h, wB_g2, hw);
    gcn_self  <<<gNH,    TB, 0, stream>>>(hw, dinv, gc2_b, agg);
    gcn_agg   <<<gAgg,   TB, 0, stream>>>(hw, ei, dinv, agg);
    relu_copy <<<gNH,    TB, 0, stream>>>(agg, h);

    fc_kernel<<<gTiles, TB, 0, stream>>>(h, wB_fc, fc_b, out);
}
